// EncCLSTMBlock_54485955117624
// MI455X (gfx1250) — matrix operand order repaired and checked by reading, not compiled, not run on hardware
//
#include <hip/hip_runtime.h>

// ---------------------------------------------------------------------------
// Types
// ---------------------------------------------------------------------------
typedef __bf16 bf16;
typedef __attribute__((ext_vector_type(16))) __bf16 v16bf;
typedef __attribute__((ext_vector_type(8)))  float  v8f;
typedef __attribute__((ext_vector_type(4)))  unsigned u32x4;

__device__ __forceinline__ bf16 f2bf(float f) {
  unsigned u = __builtin_bit_cast(unsigned, f);
  unsigned r = u + 0x7FFFu + ((u >> 16) & 1u);           // round-to-nearest-even
  unsigned short h = (unsigned short)(r >> 16);
  return __builtin_bit_cast(bf16, h);
}
__device__ __forceinline__ float bf2f(bf16 b) {
  unsigned short h = __builtin_bit_cast(unsigned short, b);
  unsigned u = ((unsigned)h) << 16;
  return __builtin_bit_cast(float, u);
}
__device__ __forceinline__ float hsig(float x) {
  return fminf(fmaxf(0.2f * x + 0.5f, 0.0f), 1.0f);
}

#define ROWB 80   // 32 bf16 = 64B payload + 16B pad -> conflict-free, 16B aligned
#define TM   128  // M tile rows (4 waves x 32-row strips)

// ---------------------------------------------------------------------------
// Implicit-GEMM ConvLSTM gate kernel with tap-padded K layout:
//   K index k = tap*Kin + ci, tap = ky*3+kx (9 taps), Kin = Cinp + C,
//   ci in [0, Cinp)   -> channel ci of padded x_t frame
//   ci in [Cinp, Kin) -> channel ci-Cinp of h_{t-1}
// Cinp and C are multiples of 16, so every 16-aligned 16-element K segment is
// one contiguous, 16B-aligned channel run of a single tensor -> b128 gather.
// Workgroup: 4 wave32, 128x64 tile; each wave owns a 32-row strip:
// 2 A-frags x 4 B-frags = 8 WMMAs per K-step (B frags reused across strips).
// ---------------------------------------------------------------------------
__global__ __launch_bounds__(128)
void gate_gemm(const bf16* __restrict__ xin, const bf16* __restrict__ hin,
               const bf16* __restrict__ wpack, float* __restrict__ g,
               int M, int H, int W, int Cinp, int C, int N, int Ktot, int Kin)
{
  __shared__ alignas(16) char As[TM * ROWB];
  __shared__ alignas(16) char Bs[64 * ROWB];

  const int m0   = blockIdx.x * TM;
  const int n0   = blockIdx.y * 64;
  const int tid  = threadIdx.x;
  const int wave = tid >> 5;
  const int lane = tid & 31;
  const int HW   = H * W;

  v8f acc[2][4] = {};

  // staging: thread stages K half `ah` of rows ar and ar+64
  const int ar = tid >> 1;
  const int ah = (tid & 1) << 4;
  const int am1 = m0 + ar;
  const int am2 = am1 + 64;
  const bool mv1 = am1 < M;
  const bool mv2 = am2 < M;
  const int ab1 = am1 / HW; const int ar1 = am1 - ab1 * HW;
  const int ay1 = ar1 / W;  const int ax1 = ar1 - ay1 * W;
  const int ab2 = am2 / HW; const int ar2 = am2 - ab2 * HW;
  const int ay2 = ar2 / W;  const int ax2 = ar2 - ay2 * W;

  // incremental K decode: k0 = kb*32 + ah = tap*Kin + ci0 (Kin >= 32 -> <=1 wrap)
  int ci0 = ah;
  int ky = 0, kx = 0;

  const int Kblocks = (Ktot + 31) >> 5;
  for (int kb = 0; kb < Kblocks; ++kb) {
    __syncthreads();
    // ---- stage A segments: 16 contiguous channels via b128 loads, 2 rows ----
    {
      const int k0 = (kb << 5) + ah;
      const bool kv = k0 < Ktot;
      const int dy = ky - 1, dx = kx - 1;
      u32x4 lo1 = {}, hi1 = {}, lo2 = {}, hi2 = {};
      if (kv) {
        const int iy1 = ay1 + dy, ix1 = ax1 + dx;
        if (mv1 && (unsigned)iy1 < (unsigned)H && (unsigned)ix1 < (unsigned)W) {
          const long base = (long)ab1 * HW + (long)iy1 * W + ix1;
          const bf16* src = (ci0 < Cinp) ? (xin + base * Cinp + ci0)
                                         : (hin + base * C + (ci0 - Cinp));
          lo1 = *(const u32x4*)src;
          hi1 = *(const u32x4*)(src + 8);
        }
        const int iy2 = ay2 + dy, ix2 = ax2 + dx;
        if (mv2 && (unsigned)iy2 < (unsigned)H && (unsigned)ix2 < (unsigned)W) {
          const long base = (long)ab2 * HW + (long)iy2 * W + ix2;
          const bf16* src = (ci0 < Cinp) ? (xin + base * Cinp + ci0)
                                         : (hin + base * C + (ci0 - Cinp));
          lo2 = *(const u32x4*)src;
          hi2 = *(const u32x4*)(src + 8);
        }
      }
      bf16* dst1 = (bf16*)(As + ar * ROWB) + ah;
      *(u32x4*)dst1       = lo1;
      *(u32x4*)(dst1 + 8) = hi1;
      bf16* dst2 = (bf16*)(As + (ar + 64) * ROWB) + ah;
      *(u32x4*)dst2       = lo2;
      *(u32x4*)(dst2 + 8) = hi2;
      // advance decode by 32 k
      ci0 += 32;
      if (ci0 >= Kin) {
        ci0 -= Kin;
        if (++kx == 3) { kx = 0; ++ky; }
      }
    }
    // ---- stage B tile (pre-packed weights): 64 cols x 32 k ----
    {
      const int col = tid >> 1;
      const int n   = n0 + col;
      const u32x4 w = *(const u32x4*)(wpack + (((long)kb * N + n) << 5) + ah);
      const u32x4 w2 = *(const u32x4*)(wpack + (((long)kb * N + n) << 5) + ah + 8);
      *(u32x4*)((bf16*)(Bs + col * ROWB) + ah) = w;
      *(u32x4*)((bf16*)(Bs + col * ROWB) + ah + 8) = w2;
    }
    __syncthreads();
    // ---- WMMA: 2 A-frags + 4 B-frags preloaded, 8 back-to-back WMMAs ----
    {
      union Frag { v16bf v; u32x4 q[2]; };
      Frag af[2], bfr[4];
      // A 16x32 bf16 layout (ISA 7.12.2): lanes 0-15 hold K0-7 & K16-23
      const int  aoff  = (lane < 16) ? 0 : 16;
      const char* ab0 = As + ((wave << 5) + (lane & 15)) * ROWB + aoff;
      af[0].q[0] = *(const u32x4*)(ab0);
      af[0].q[1] = *(const u32x4*)(ab0 + 32);
      af[1].q[0] = *(const u32x4*)(ab0 + 16 * ROWB);
      af[1].q[1] = *(const u32x4*)(ab0 + 16 * ROWB + 32);
      // B 32x16 bf16: lane = column, lanes 0-15 K0-7 & K16-23, lanes 16-31 K8-15 & K24-31
      const int  boff  = (lane < 16) ? 0 : 16;
      const char* bb = Bs + (lane & 15) * ROWB + boff;
      #pragma unroll
      for (int j = 0; j < 4; ++j) {
        bfr[j].q[0] = *(const u32x4*)(bb + j * 16 * ROWB);
        bfr[j].q[1] = *(const u32x4*)(bb + j * 16 * ROWB + 32);
      }
      #pragma unroll
      for (int i = 0; i < 2; ++i) {
        #pragma unroll
        for (int j = 0; j < 4; ++j) {
          acc[i][j] = __builtin_amdgcn_wmma_f32_16x16x32_bf16(
              false, af[i].v, false, bfr[j].v, (short)0, acc[i][j], false, false);
        }
      }
    }
  }
  // ---- epilogue: D layout VGPR r -> M=r (lanes 0-15) / M=r+8 (lanes 16-31) ----
  const int ncol = n0 + (lane & 15);
  #pragma unroll
  for (int i = 0; i < 2; ++i) {
    const int mbase = m0 + (wave << 5) + (i << 4) + ((lane >> 4) << 3);
    #pragma unroll
    for (int r = 0; r < 8; ++r) {
      const int m = mbase + r;
      if (m < M) {
        float* go = g + (long)m * N + ncol;
        go[0]  = acc[i][0][r];
        go[16] = acc[i][1][r];
        go[32] = acc[i][2][r];
        go[48] = acc[i][3][r];
      }
    }
  }
}

// ---------------------------------------------------------------------------
// LSTM cell update: i,f,c~,o gates -> c, h (h also into hs[t] slice)
// ---------------------------------------------------------------------------
__global__ void gate_update(const float* __restrict__ g, const float* __restrict__ bias,
                            float* __restrict__ c, bf16* __restrict__ h,
                            bf16* __restrict__ hs_t, long MC, int C)
{
  const long idx = (long)blockIdx.x * blockDim.x + threadIdx.x;
  if (idx >= MC) return;
  const long m = idx / C;
  const int  ch = (int)(idx - m * C);
  const float* gr = g + m * 4 * C;
  const float gi = gr[ch]         + bias[ch];
  const float gf = gr[C + ch]     + bias[C + ch];
  const float gc = gr[2 * C + ch] + bias[2 * C + ch];
  const float go = gr[3 * C + ch] + bias[3 * C + ch];
  const float i = hsig(gi);
  const float f = hsig(gf);
  const float o = hsig(go);
  const float cn = f * c[idx] + i * tanhf(gc);
  const float hn = o * tanhf(cn);
  c[idx] = cn;
  const bf16 hb = f2bf(hn);
  h[idx] = hb;
  hs_t[idx] = hb;
}

// ---------------------------------------------------------------------------
// Weight packing: W(3,3,Cin,4C) + U(3,3,C,4C) f32 -> bf16 [kblock][n][32]
// Tap-padded K: k = tap*Kin + ci; ci<Cin -> W, Cinp<=ci<Cinp+C -> U, else 0
// ---------------------------------------------------------------------------
__global__ void pack_weights(const float* __restrict__ Wg, const float* __restrict__ Ug,
                             bf16* __restrict__ wpack, int Cin, int Cinp, int C,
                             int N, int Ktot, int Kin, long total)
{
  const long idx = (long)blockIdx.x * blockDim.x + threadIdx.x;
  if (idx >= total) return;
  const int  kk = (int)(idx & 31);
  const long rest = idx >> 5;
  const int  n  = (int)(rest % N);
  const int  kb = (int)(rest / N);
  const int  k  = kb * 32 + kk;
  float v = 0.0f;
  if (k < Ktot) {
    const int tap = k / Kin;
    const int ci  = k - tap * Kin;
    if (ci < Cin)
      v = Wg[((long)tap * Cin + ci) * N + n];
    else if (ci >= Cinp && ci < Cinp + C)
      v = Ug[((long)tap * C + (ci - Cinp)) * N + n];
  }
  wpack[idx] = f2bf(v);
}

// ---------------------------------------------------------------------------
// Input convert: x f32 (B,T,H,W,Cin) -> bf16 [T][B][H][W][Cinp], zero-padded
// ---------------------------------------------------------------------------
__global__ void convert_input(const float* __restrict__ in, bf16* __restrict__ out,
                              int B, int T, long HW, int Cin, int Cinp, long total)
{
  const long idx = (long)blockIdx.x * blockDim.x + threadIdx.x;
  if (idx >= total) return;
  const int  ch = (int)(idx % Cinp);
  const long p  = idx / Cinp;          // (t*B + b)*HW + s
  const long s  = p % HW;
  const long r  = p / HW;
  const int  b  = (int)(r % B);
  const int  t  = (int)(r / B);
  float v = 0.0f;
  if (ch < Cin) v = in[(((long)b * T + t) * HW + s) * Cin + ch];
  out[idx] = f2bf(v);
}

// ---------------------------------------------------------------------------
// Fused BatchNorm (inference) + 3x3/3 SAME avg-pool (count excludes padding)
// hs layout [T][B][H][W][C]; bf16 out -> next layer input (same layout);
// f32 out (last layer) -> (B,T,Hp,Wp,C) order for d_out.
// ---------------------------------------------------------------------------
__global__ void bnpool(const bf16* __restrict__ hs,
                       const float* __restrict__ gamma, const float* __restrict__ beta,
                       const float* __restrict__ mean,  const float* __restrict__ var,
                       bf16* __restrict__ outb, float* __restrict__ outf,
                       int B, int T, int H, int W, int C, int Hp, int Wp,
                       int padh, int padw, long total)
{
  const long idx = (long)blockIdx.x * blockDim.x + threadIdx.x;
  if (idx >= total) return;
  const int ch = (int)(idx % C);
  long r = idx / C;
  const int ox = (int)(r % Wp); r /= Wp;
  const int oy = (int)(r % Hp); r /= Hp;
  const int b  = (int)(r % B);
  const int t  = (int)(r / B);
  const float sc = gamma[ch] * rsqrtf(var[ch] + 1e-3f);
  const float sh = beta[ch] - mean[ch] * sc;
  float s = 0.0f;
  int cnt = 0;
  #pragma unroll
  for (int dy = 0; dy < 3; ++dy) {
    const int iy = oy * 3 - padh + dy;
    if (iy < 0 || iy >= H) continue;
    #pragma unroll
    for (int dx = 0; dx < 3; ++dx) {
      const int ix = ox * 3 - padw + dx;
      if (ix < 0 || ix >= W) continue;
      const long src = ((((long)t * B + b) * H + iy) * W + ix) * (long)C + ch;
      s += bf2f(hs[src]) * sc + sh;
      ++cnt;
    }
  }
  const float v = s / (float)cnt;
  if (outf)
    outf[(((long)b * T + t) * Hp * Wp + (long)oy * Wp + ox) * C + ch] = v;
  else
    outb[idx] = f2bf(v);
}

// ---------------------------------------------------------------------------
// Host orchestration
// ---------------------------------------------------------------------------
static inline unsigned nblk(long total, int thr) {
  return (unsigned)((total + thr - 1) / thr);
}

extern "C" void kernel_launch(void* const* d_in, const int* in_sizes, int n_in,
                              void* d_out, int out_size, void* d_ws, size_t ws_size,
                              hipStream_t stream)
{
  (void)in_sizes; (void)n_in; (void)out_size; (void)ws_size;
  const int B = 4, T = 16;
  static const int Hs[5]    = {192, 64, 22, 8, 3};
  static const int Hp[5]    = {64, 22, 8, 3, 1};
  static const int padv[5]  = {0, 1, 1, 0, 0};
  static const int Cins[5]  = {3, 16, 32, 64, 64};   // real input channels
  static const int Cinp[5]  = {16, 16, 32, 64, 64};  // padded to multiple of 16
  static const int Cs[5]    = {16, 32, 64, 64, 64};

  // setup_inputs() dict order: x, then per layer {W, U, b, gamma, beta, mean, var}
  const float* x = (const float*)d_in[0];
  const float *Wg[5], *Ug[5], *bg[5], *gam[5], *bet[5], *mea[5], *var[5];
  for (int l = 0; l < 5; ++l) {
    Wg[l]  = (const float*)d_in[1 + 7 * l + 0];
    Ug[l]  = (const float*)d_in[1 + 7 * l + 1];
    bg[l]  = (const float*)d_in[1 + 7 * l + 2];
    gam[l] = (const float*)d_in[1 + 7 * l + 3];
    bet[l] = (const float*)d_in[1 + 7 * l + 4];
    mea[l] = (const float*)d_in[1 + 7 * l + 5];
    var[l] = (const float*)d_in[1 + 7 * l + 6];
  }

  // bump-allocate scratch (~216 MB; hot set stays L2-resident)
  char* ws = (char*)d_ws;
  size_t off = 0;
  auto alloc = [&](size_t bytes) -> char* {
    char* p = ws + off;
    off = (off + bytes + 255) & ~(size_t)255;
    return p;
  };

  bf16* xin[5];
  for (int l = 0; l < 5; ++l)
    xin[l] = (bf16*)alloc((size_t)T * B * Hs[l] * Hs[l] * Cinp[l] * sizeof(bf16));

  bf16* wpk[5];
  int Kin[5], Ktot[5], Kb[5], Nn[5];
  for (int l = 0; l < 5; ++l) {
    Kin[l]  = Cinp[l] + Cs[l];
    Ktot[l] = 9 * Kin[l];
    Kb[l]   = (Ktot[l] + 31) / 32;
    Nn[l]   = 4 * Cs[l];
    wpk[l]  = (bf16*)alloc((size_t)Kb[l] * Nn[l] * 32 * sizeof(bf16));
  }
  bf16*  hs   = (bf16*)alloc((size_t)T * B * 192 * 192 * 16 * sizeof(bf16));
  float* g    = (float*)alloc((size_t)B * 192 * 192 * 64 * sizeof(float));
  bf16*  hbuf = (bf16*)alloc((size_t)B * 192 * 192 * 16 * sizeof(bf16));
  float* cbuf = (float*)alloc((size_t)B * 192 * 192 * 16 * sizeof(float));

  // input -> bf16, (B,T) -> (T,B) major, channel-padded to Cinp[0]
  {
    const long HW = (long)192 * 192;
    const long tot = (long)T * B * HW * Cinp[0];
    convert_input<<<dim3(nblk(tot, 256)), 256, 0, stream>>>(
        x, xin[0], B, T, HW, Cins[0], Cinp[0], tot);
  }

  for (int l = 0; l < 5; ++l) {
    const int H = Hs[l], Wd = Hs[l], C = Cs[l], N = Nn[l];
    const int M = B * H * Wd;

    {
      const long tot = (long)Kb[l] * N * 32;
      pack_weights<<<dim3(nblk(tot, 256)), 256, 0, stream>>>(
          Wg[l], Ug[l], wpk[l], Cins[l], Cinp[l], C, N, Ktot[l], Kin[l], tot);
    }
    hipMemsetAsync(hbuf, 0, (size_t)M * C * sizeof(bf16), stream);
    hipMemsetAsync(cbuf, 0, (size_t)M * C * sizeof(float), stream);

    for (int t = 0; t < T; ++t) {
      const bf16* xt = xin[l] + (size_t)t * M * Cinp[l];
      dim3 grid((M + TM - 1) / TM, N / 64);
      gate_gemm<<<grid, 128, 0, stream>>>(xt, hbuf, wpk[l], g,
                                          M, H, Wd, Cinp[l], C, N, Ktot[l], Kin[l]);
      const long tot = (long)M * C;
      gate_update<<<dim3(nblk(tot, 256)), 256, 0, stream>>>(
          g, bg[l], cbuf, hbuf, hs + (size_t)t * M * C, tot, C);
    }

    {
      const int hp = Hp[l], pad = padv[l];
      const long tot = (long)T * B * hp * hp * C;
      bf16*  outb = (l < 4) ? xin[l + 1] : nullptr;
      float* outf = (l == 4) ? (float*)d_out : nullptr;
      bnpool<<<dim3(nblk(tot, 256)), 256, 0, stream>>>(
          hs, gam[l], bet[l], mea[l], var[l], outb, outf,
          B, T, H, Wd, C, hp, hp, pad, pad, tot);
    }
  }
}
